// EXACTLoss_71055938945755
// MI455X (gfx1250) — compile-verified, matrix operand order checked by reading, not run on hardware
//
#include <hip/hip_runtime.h>

typedef __attribute__((ext_vector_type(16))) _Float16 v16h;
typedef __attribute__((ext_vector_type(8)))  float    v8f;

#define NB 8192
#define NC 1024
#define NS 16
#define BLK_ROWS 8
#define NBLOCKS (NB / BLK_ROWS)      /* 1024 main blocks  */
#define STAT_BLOCKS 1024

// t_k = ndtri((k+0.5)/16), midpoint quantile nodes (accurate to ~2e-6)
__device__ __constant__ float T_NODES[16] = {
    -1.8627320f, -1.3180140f, -1.0099900f, -0.7764200f,
    -0.5791310f, -0.4022490f, -0.2372020f, -0.0784130f,
     0.0784130f,  0.2372020f,  0.4022490f,  0.5791310f,
     0.7764200f,  1.0099900f,  1.3180140f,  1.8627320f };

// Branch-free log Phi(x).  Phi(x) = 0.5*erfc(-x/sqrt(2)).
// log1p(Phi-1) is the accurate path where it matters (Phi near 1, i.e. terms
// near 0 that dominate the product); for Phi -> 0 the absolute error (or the
// -inf limit) only affects terms <= -14 which force exp(sum) -> 0 regardless.
__device__ __forceinline__ float log_ndtr(float x) {
    const float INV_SQRT2 = 0.70710678118654752440f;
    float phi = 0.5f * erfcf(-x * INV_SQRT2);
    return log1pf(phi - 1.0f);
}

// ---------------- Pass 1a: per-block partial sum / sumsq --------------------
__global__ __launch_bounds__(256) void stats_partial_kernel(
        const float4* __restrict__ x, float2* __restrict__ part, int n4) {
    const int tid = threadIdx.x;
    float s = 0.0f, ss = 0.0f;
    for (int i = blockIdx.x * 256 + tid; i < n4; i += STAT_BLOCKS * 256) {
        float4 v = x[i];
        s  += (v.x + v.y) + (v.z + v.w);
        ss += (v.x * v.x + v.y * v.y) + (v.z * v.z + v.w * v.w);
    }
    __shared__ float as_[256], bs_[256];
    as_[tid] = s; bs_[tid] = ss;
    __syncthreads();
    for (int off = 128; off > 0; off >>= 1) {
        if (tid < off) { as_[tid] += as_[tid + off]; bs_[tid] += bs_[tid + off]; }
        __syncthreads();
    }
    if (tid == 0) part[blockIdx.x] = make_float2(as_[0], bs_[0]);
}

// ---------------- Pass 1b: finalize scale = rsqrt(var_ddof1) / T ------------
__global__ __launch_bounds__(256) void stats_final_kernel(
        const float2* __restrict__ part, const float* __restrict__ temperature,
        float* __restrict__ scale_out) {
    const int tid = threadIdx.x;
    float s = 0.0f, ss = 0.0f;
    for (int i = tid; i < STAT_BLOCKS; i += 256) {
        float2 p = part[i]; s += p.x; ss += p.y;
    }
    __shared__ float as_[256], bs_[256];
    as_[tid] = s; bs_[tid] = ss;
    __syncthreads();
    for (int off = 128; off > 0; off >>= 1) {
        if (tid < off) { as_[tid] += as_[tid + off]; bs_[tid] += bs_[tid + off]; }
        __syncthreads();
    }
    if (tid == 0) {
        const float n = (float)NB * (float)NC;
        float mean = as_[0] / n;
        float var  = (bs_[0] - n * mean * mean) / (n - 1.0f);  // ddof=1
        scale_out[0] = rsqrtf(var) / temperature[0];
    }
}

// ---------------- Pass 2: main kernel (WMMA reduction over classes) ---------
// Block = 256 threads = 8 wave32 waves; wave w owns batch row blk*8+w.
// Per wave: A(16x32) f16 tile of masked logcdf values, B = ones(32x16),
// f32 accumulator carries the 16 per-sample column sums across 32 chunks.
__global__ __launch_bounds__(256) void exact_main_kernel(
        const float* __restrict__ logits, const int* __restrict__ labels,
        const float* __restrict__ scale_p, float* __restrict__ block_prob) {
    __shared__ __align__(16) float rowbuf[BLK_ROWS * NC];   // 32 KB
    __shared__ float wprob[BLK_ROWS];

    const int tid = threadIdx.x;
    const int blk = blockIdx.x;

    // Stage 8 rows (32 KB) into LDS: each logit read from HBM exactly once.
    {
        const float4* src = (const float4*)(logits + (size_t)blk * BLK_ROWS * NC);
        float4* dst = (float4*)rowbuf;
        #pragma unroll
        for (int it = 0; it < (BLK_ROWS * NC / 4) / 256; ++it)
            dst[tid + it * 256] = src[tid + it * 256];
    }
    __syncthreads();

    const float scale = scale_p[0];
    const int wave = tid >> 5;          // wave32
    const int lane = tid & 31;
    const int row  = blk * BLK_ROWS + wave;
    const float* __restrict__ r = rowbuf + wave * NC;

    const int   y   = labels[row];
    const float tgt = r[y];
    const float t   = T_NODES[lane & 15];   // A-matrix row M = lane%16
    const int   jhi = (lane & 16) >> 1;     // lanes 16-31 cover K+8 halves
    const float c0  = tgt * scale + t;      // x = c0 - val*scale  (one FMA)

    v16h bones;
    #pragma unroll
    for (int e = 0; e < 16; ++e) bones[e] = (_Float16)1.0f;

    v8f acc = {};
    #pragma unroll 2
    for (int c = 0; c < NC / 32; ++c) {
        const int base = c * 32;
        // lane needs columns base+jhi+{0..7} and base+jhi+{16..23}
        const float4* r4 = (const float4*)(r + base + jhi);
        float4 u0 = r4[0], u1 = r4[1], u2 = r4[4], u3 = r4[5];
        float vals[16] = { u0.x, u0.y, u0.z, u0.w,  u1.x, u1.y, u1.z, u1.w,
                           u2.x, u2.y, u2.z, u2.w,  u3.x, u3.y, u3.z, u3.w };
        v16h a;
        #pragma unroll
        for (int e = 0; e < 16; ++e) {
            // ISA 16-bit A layout: K = e + 8*(e>=8) + 8*(lane>=16)
            const int j = base + jhi + e + ((e >= 8) ? 8 : 0);
            float lc = log_ndtr(c0 - vals[e] * scale);  // unconditional
            lc = (j == y) ? 0.0f : lc;                  // alt_mask: v_cndmask
            a[e] = (_Float16)lc;
        }
        acc = __builtin_amdgcn_wmma_f32_16x16x32_f16(
                  false, a, false, bones, (short)0, acc, false, false);
    }

    // C/D layout: VGPR i holds M=i (lanes 0-15) / M=i+8 (lanes 16-31);
    // all 16 columns identical (B = ones). Sum exp over this lane's 8 samples,
    // then combine the two half-wave groups.
    float ps = 0.0f;
    #pragma unroll
    for (int i = 0; i < 8; ++i) ps += expf(acc[i]);
    ps += __shfl_xor(ps, 16, 32);

    if (lane == 0) wprob[wave] = ps * (1.0f / (float)NS);
    __syncthreads();
    if (tid == 0) {
        float s = 0.0f;
        #pragma unroll
        for (int w = 0; w < BLK_ROWS; ++w) s += wprob[w];
        block_prob[blk] = s;
    }
}

// ---------------- Pass 3: final reduction -> loss ---------------------------
__global__ __launch_bounds__(256) void final_loss_kernel(
        const float* __restrict__ part, float* __restrict__ out) {
    const int tid = threadIdx.x;
    float s = 0.0f;
    for (int i = tid; i < NBLOCKS; i += 256) s += part[i];
    __shared__ float as_[256];
    as_[tid] = s;
    __syncthreads();
    for (int off = 128; off > 0; off >>= 1) {
        if (tid < off) as_[tid] += as_[tid + off];
        __syncthreads();
    }
    if (tid == 0) out[0] = 1.0f - as_[0] * (1.0f / (float)NB);
}

extern "C" void kernel_launch(void* const* d_in, const int* in_sizes, int n_in,
                              void* d_out, int out_size, void* d_ws, size_t ws_size,
                              hipStream_t stream) {
    (void)in_sizes; (void)n_in; (void)out_size; (void)ws_size;
    const float* logits      = (const float*)d_in[0];
    const int*   labels      = (const int*)d_in[1];
    const float* temperature = (const float*)d_in[2];
    float*       out         = (float*)d_out;

    char* ws = (char*)d_ws;
    float*  scale_p   = (float*)ws;                                   // 16 B slot
    float2* stat_part = (float2*)(ws + 16);                           // 8 KB
    float*  prob_part = (float*)(ws + 16 + sizeof(float2) * STAT_BLOCKS); // 4 KB

    stats_partial_kernel<<<STAT_BLOCKS, 256, 0, stream>>>(
        (const float4*)logits, stat_part, NB * NC / 4);
    stats_final_kernel<<<1, 256, 0, stream>>>(stat_part, temperature, scale_p);
    exact_main_kernel<<<NBLOCKS, 256, 0, stream>>>(logits, labels, scale_p, prob_part);
    final_loss_kernel<<<1, 256, 0, stream>>>(prob_part, out);
}
